// GeneralGraphNetwork_49246095016469
// MI455X (gfx1250) — matrix operand order repaired and checked by reading, not compiled, not run on hardware
//
#include <hip/hip_runtime.h>
#include <hip/hip_bf16.h>

#define NND 50000
#define NED 800000
#define NB  4
#define HID 64
#define LSTRIDE 72          // 16-row LDS scratch stride (halfs), conflict-free & 16B aligned
#define WVB 8               // waves per block
#define BT  256             // threads per block

typedef __attribute__((ext_vector_type(16))) _Float16 v16h;
typedef __attribute__((ext_vector_type(8)))  _Float16 v8h;
typedef __attribute__((ext_vector_type(8)))  float    v8f;

// ---------------- device helpers ----------------

__device__ __forceinline__ v8f wmma16(v16h a, v16h b, v8f c) {
  return __builtin_amdgcn_wmma_f32_16x16x32_f16(false, a, false, b, (short)0, c,
                                                false, false);
}

// Build 16x32 f16 A-fragment from a 64-float (or 32-float-chunk) fp32 row.
// lane m = lane&15 owns row m; half8 = 8*(lane>=16).
// element i -> k = (i&8)*2 + (i&7) + half8  => two contiguous 8-float segments.
__device__ __forceinline__ v16h a_from_global(const float* p, int half8) {
  const float4* q0 = (const float4*)(p + half8);
  const float4* q1 = (const float4*)(p + 16 + half8);
  float4 x0 = q0[0], x1 = q0[1], y0 = q1[0], y1 = q1[1];
  v16h a;
  a[0]=(_Float16)x0.x; a[1]=(_Float16)x0.y; a[2]=(_Float16)x0.z; a[3]=(_Float16)x0.w;
  a[4]=(_Float16)x1.x; a[5]=(_Float16)x1.y; a[6]=(_Float16)x1.z; a[7]=(_Float16)x1.w;
  a[8]=(_Float16)y0.x; a[9]=(_Float16)y0.y; a[10]=(_Float16)y0.z; a[11]=(_Float16)y0.w;
  a[12]=(_Float16)y1.x; a[13]=(_Float16)y1.y; a[14]=(_Float16)y1.z; a[15]=(_Float16)y1.w;
  return a;
}

__device__ __forceinline__ v16h a_from_global_scaled(const float* p, int half8, float s) {
  const float4* q0 = (const float4*)(p + half8);
  const float4* q1 = (const float4*)(p + 16 + half8);
  float4 x0 = q0[0], x1 = q0[1], y0 = q1[0], y1 = q1[1];
  v16h a;
  a[0]=(_Float16)(x0.x*s); a[1]=(_Float16)(x0.y*s); a[2]=(_Float16)(x0.z*s); a[3]=(_Float16)(x0.w*s);
  a[4]=(_Float16)(x1.x*s); a[5]=(_Float16)(x1.y*s); a[6]=(_Float16)(x1.z*s); a[7]=(_Float16)(x1.w*s);
  a[8]=(_Float16)(y0.x*s); a[9]=(_Float16)(y0.y*s); a[10]=(_Float16)(y0.z*s); a[11]=(_Float16)(y0.w*s);
  a[12]=(_Float16)(y1.x*s); a[13]=(_Float16)(y1.y*s); a[14]=(_Float16)(y1.z*s); a[15]=(_Float16)(y1.w*s);
  return a;
}

__device__ __forceinline__ v16h a_from_lds(const _Float16* row, int ck, int half8) {
  v8h lo = *(const v8h*)(row + ck + half8);
  v8h hi = *(const v8h*)(row + ck + 16 + half8);
  v16h a;
#pragma unroll
  for (int i = 0; i < 8; ++i) { a[i] = lo[i]; a[8 + i] = hi[i]; }
  return a;
}

// Pre-swizzled weights: layout [((c*NT + t)*32 + kk)*16 + nn], B lane = kk, elem = nn.
__device__ __forceinline__ v16h b_from_swz(const _Float16* w, int c, int t, int nt, int lane) {
  return *(const v16h*)(w + ((size_t)((c * nt + t) * 32 + lane) << 4));
}

// Store C-layout result (+bias, optional relu) as f16 into per-wave LDS [16][LSTRIDE]
__device__ __forceinline__ void c_to_lds(_Float16* lds, const v8f acc[4],
                                         const float* bias, bool relu, int lane) {
  int nl = lane & 15, mh = (lane >> 4) * 8;
#pragma unroll
  for (int t = 0; t < 4; ++t) {
    float bv = bias[t * 16 + nl];
#pragma unroll
    for (int r = 0; r < 8; ++r) {
      float v = acc[t][r] + bv;
      if (relu) v = v > 0.f ? v : 0.f;
      lds[(mh + r) * LSTRIDE + t * 16 + nl] = (_Float16)v;
    }
  }
}

// second Linear (K=64 from LDS-transposed hidden) into 4 N tiles
__device__ __forceinline__ void stage2_4t(const _Float16* myrow, const _Float16* w2,
                                          v8f acc[4], int half8, int lane) {
#pragma unroll
  for (int c = 0; c < 2; ++c) {
    v16h a = a_from_lds(myrow, c * 32, half8);
#pragma unroll
    for (int t = 0; t < 4; ++t) acc[t] = wmma16(a, b_from_swz(w2, c, t, 4, lane), acc[t]);
  }
}

__device__ __forceinline__ void zero4(v8f acc[4]) {
  v8f z = {};
#pragma unroll
  for (int t = 0; t < 4; ++t) acc[t] = z;
}

// ---------------- utility kernels ----------------

__global__ void zero_kernel(float* p, long n) {
  long i = (long)blockIdx.x * blockDim.x + threadIdx.x;
  long stride = (long)gridDim.x * blockDim.x;
  for (; i < n; i += stride) p[i] = 0.f;
}

__global__ void count_kernel(const int* __restrict__ col, float* __restrict__ cnt, int E) {
  int i = blockIdx.x * blockDim.x + threadIdx.x;
  if (i < E) atomicAdd(&cnt[col[i]], 1.0f);
}

// convert + pre-swizzle weight [K][Nout] fp32 -> f16 fragment layout (zero padded)
__global__ void swizzle_w_kernel(const float* __restrict__ W, _Float16* __restrict__ out,
                                 int K, int Nout, int NT) {
  int nch = (K + 31) / 32;
  int total = nch * NT * 512;
  for (int idx = blockIdx.x * blockDim.x + threadIdx.x; idx < total;
       idx += gridDim.x * blockDim.x) {
    int nn = idx & 15;
    int kk = (idx >> 4) & 31;
    int ct = idx >> 9;
    int t = ct % NT, c = ct / NT;
    int k = c * 32 + ((nn & 8) << 1) + (nn & 7) + ((kk >> 4) << 3), n = t * 16 + (kk & 15);
    float v = (k < K && n < Nout) ? W[(size_t)k * Nout + n] : 0.f;
    out[idx] = (_Float16)v;
  }
}

// global encoder: only 4 rows -> plain VALU
__global__ void glob_encoder_kernel(const float* __restrict__ u, const float* __restrict__ W1,
                                    const float* __restrict__ b1, const float* __restrict__ W2,
                                    const float* __restrict__ b2, float* __restrict__ g) {
  __shared__ float hid[NB][HID];
  int n = threadIdx.x;  // 64 threads
  for (int b = 0; b < NB; ++b) {
    float s = b1[n];
    for (int k = 0; k < 4; ++k) s += u[b * 4 + k] * W1[k * HID + n];
    hid[b][n] = s > 0.f ? s : 0.f;
  }
  __syncthreads();
  for (int b = 0; b < NB; ++b) {
    float s = b2[n];
    for (int j = 0; j < HID; ++j) s += hid[b][j] * W2[j * HID + n];
    g[b * HID + n] = s;
  }
}

// ---------------- WMMA MLP kernels ----------------

// MLP2 encoder with small fan-in (<=32): dst[rows][64] = relu(src@W1+b1)@W2+b2
__global__ void encoder_kernel(const float* __restrict__ src, int ntiles, int Kin,
                               const _Float16* __restrict__ w1, const float* __restrict__ b1,
                               const _Float16* __restrict__ w2, const float* __restrict__ b2,
                               float* __restrict__ dst) {
  __shared__ __align__(32) _Float16 scr[WVB][16 * LSTRIDE];
  int lane = threadIdx.x & 31, wv = threadIdx.x >> 5;
  int tile = blockIdx.x * WVB + wv;
  if (tile >= ntiles) return;
  int m = lane & 15, half8 = (lane >> 4) * 8;
  const float* p = src + (size_t)(tile * 16 + m) * Kin;
  v16h a;
#pragma unroll
  for (int i = 0; i < 16; ++i) {
    int k = ((i & 8) << 1) + (i & 7) + half8;
    a[i] = (k < Kin) ? (_Float16)p[k] : (_Float16)0.f;
  }
  v8f acc[4]; zero4(acc);
#pragma unroll
  for (int t = 0; t < 4; ++t) acc[t] = wmma16(a, b_from_swz(w1, 0, t, 4, lane), acc[t]);
  _Float16* my = scr[wv];
  c_to_lds(my, acc, b1, true, lane);
  v8f acc2[4]; zero4(acc2);
  const _Float16* myrow = my + (size_t)m * LSTRIDE;
  stage2_4t(myrow, w2, acc2, half8, lane);
  int nl = lane & 15, mh = (lane >> 4) * 8;
#pragma unroll
  for (int t = 0; t < 4; ++t) {
    float bv = b2[t * 16 + nl];
#pragma unroll
    for (int r = 0; r < 8; ++r)
      dst[(size_t)(tile * 16 + mh + r) * HID + t * 16 + nl] = acc2[t][r] + bv;
  }
}

// fused per-layer edge work: e_new = MLP(cat(h[row],h[col],e,g[eb])); e += e_new;
// m = MLP(cat(h[row], e_new)); agg[col] += m  (atomic)
__global__ void edge_layer_kernel(const float* __restrict__ h, float* __restrict__ e,
                                  const float* __restrict__ g, const int* __restrict__ rowI,
                                  const int* __restrict__ colI, const int* __restrict__ batch,
                                  const _Float16* __restrict__ w1e, const float* __restrict__ b1e,
                                  const _Float16* __restrict__ w2e, const float* __restrict__ b2e,
                                  const _Float16* __restrict__ w1n, const float* __restrict__ b1n,
                                  const _Float16* __restrict__ w2n, const float* __restrict__ b2n,
                                  float* __restrict__ agg) {
  __shared__ __align__(32) _Float16 scr[WVB][16 * LSTRIDE];
  int lane = threadIdx.x & 31, wv = threadIdx.x >> 5;
  int tile = blockIdx.x * WVB + wv;
  int m = lane & 15, half8 = (lane >> 4) * 8;
  int ei = tile * 16 + m;
  int rI = rowI[ei], cI = colI[ei];
  int gb = batch[rI];
  const float* hr = h + (size_t)rI * HID;
  const float* hc = h + (size_t)cI * HID;
  const float* ep = e + (size_t)ei * HID;
  const float* gp = g + gb * HID;
  const float* srcs[8] = {hr, hr + 32, hc, hc + 32, ep, ep + 32, gp, gp + 32};

  // ---- edge MLP stage 1: K=256
  v8f acc[4]; zero4(acc);
#pragma unroll
  for (int c = 0; c < 8; ++c) {
    v16h a = a_from_global(srcs[c], half8);
#pragma unroll
    for (int t = 0; t < 4; ++t) acc[t] = wmma16(a, b_from_swz(w1e, c, t, 4, lane), acc[t]);
  }
  _Float16* my = scr[wv];
  c_to_lds(my, acc, b1e, true, lane);
  const _Float16* myrow = my + (size_t)m * LSTRIDE;

  // ---- edge MLP stage 2: K=64
  v8f acc2[4]; zero4(acc2);
  stage2_4t(myrow, w2e, acc2, half8, lane);

  // residual e += e_new ; stash e_new (f16) in LDS for node1 input
  asm volatile("s_wait_loadcnt 0x0" ::: "memory");  // cross-lane RAW: A-gathers of e vs stores below
  int nl = lane & 15, mh = (lane >> 4) * 8;
#pragma unroll
  for (int t = 0; t < 4; ++t) {
    float bv = b2e[t * 16 + nl];
#pragma unroll
    for (int r = 0; r < 8; ++r) {
      float env = acc2[t][r] + bv;
      size_t idx = (size_t)(tile * 16 + mh + r) * HID + t * 16 + nl;
      e[idx] = e[idx] + env;
      my[(mh + r) * LSTRIDE + t * 16 + nl] = (_Float16)env;  // DS ops in-order: safe overwrite
    }
  }

  // ---- node1 MLP stage 1: K=128 = cat(h[row], e_new)
  v8f accm[4]; zero4(accm);
  {
    v16h a0 = a_from_global(hr, half8);
#pragma unroll
    for (int t = 0; t < 4; ++t) accm[t] = wmma16(a0, b_from_swz(w1n, 0, t, 4, lane), accm[t]);
    v16h a1 = a_from_global(hr + 32, half8);
#pragma unroll
    for (int t = 0; t < 4; ++t) accm[t] = wmma16(a1, b_from_swz(w1n, 1, t, 4, lane), accm[t]);
    v16h a2 = a_from_lds(myrow, 0, half8);
#pragma unroll
    for (int t = 0; t < 4; ++t) accm[t] = wmma16(a2, b_from_swz(w1n, 2, t, 4, lane), accm[t]);
    v16h a3 = a_from_lds(myrow, 32, half8);
#pragma unroll
    for (int t = 0; t < 4; ++t) accm[t] = wmma16(a3, b_from_swz(w1n, 3, t, 4, lane), accm[t]);
  }
  c_to_lds(my, accm, b1n, true, lane);

  // ---- node1 MLP stage 2 + scatter-add
  v8f accm2[4]; zero4(accm2);
  stage2_4t(myrow, w2n, accm2, half8, lane);
#pragma unroll
  for (int t = 0; t < 4; ++t) {
    float bv = b2n[t * 16 + nl];
#pragma unroll
    for (int r = 0; r < 8; ++r) {
      int gc = colI[tile * 16 + mh + r];
      atomicAdd(&agg[(size_t)gc * HID + t * 16 + nl], accm2[t][r] + bv);
    }
  }
}

// per-layer node work: h += MLP(cat(h, agg/cnt, g[batch]))   (K=192)
__global__ void node_layer_kernel(float* __restrict__ h, const float* __restrict__ agg,
                                  const float* __restrict__ cnt, const float* __restrict__ g,
                                  const int* __restrict__ batch,
                                  const _Float16* __restrict__ w1, const float* __restrict__ b1,
                                  const _Float16* __restrict__ w2, const float* __restrict__ b2,
                                  int ntiles) {
  __shared__ __align__(32) _Float16 scr[WVB][16 * LSTRIDE];
  int lane = threadIdx.x & 31, wv = threadIdx.x >> 5;
  int tile = blockIdx.x * WVB + wv;
  if (tile >= ntiles) return;
  int m = lane & 15, half8 = (lane >> 4) * 8;
  int ni = tile * 16 + m;
  const float* hp = h + (size_t)ni * HID;
  const float* ap = agg + (size_t)ni * HID;
  float s = 1.0f / fmaxf(cnt[ni], 1.0f);
  const float* gp = g + batch[ni] * HID;

  v8f acc[4]; zero4(acc);
  {
    v16h a;
    a = a_from_global(hp, half8);
#pragma unroll
    for (int t = 0; t < 4; ++t) acc[t] = wmma16(a, b_from_swz(w1, 0, t, 4, lane), acc[t]);
    a = a_from_global(hp + 32, half8);
#pragma unroll
    for (int t = 0; t < 4; ++t) acc[t] = wmma16(a, b_from_swz(w1, 1, t, 4, lane), acc[t]);
    a = a_from_global_scaled(ap, half8, s);
#pragma unroll
    for (int t = 0; t < 4; ++t) acc[t] = wmma16(a, b_from_swz(w1, 2, t, 4, lane), acc[t]);
    a = a_from_global_scaled(ap + 32, half8, s);
#pragma unroll
    for (int t = 0; t < 4; ++t) acc[t] = wmma16(a, b_from_swz(w1, 3, t, 4, lane), acc[t]);
    a = a_from_global(gp, half8);
#pragma unroll
    for (int t = 0; t < 4; ++t) acc[t] = wmma16(a, b_from_swz(w1, 4, t, 4, lane), acc[t]);
    a = a_from_global(gp + 32, half8);
#pragma unroll
    for (int t = 0; t < 4; ++t) acc[t] = wmma16(a, b_from_swz(w1, 5, t, 4, lane), acc[t]);
  }
  _Float16* my = scr[wv];
  c_to_lds(my, acc, b1, true, lane);
  v8f acc2[4]; zero4(acc2);
  stage2_4t(my + (size_t)m * LSTRIDE, w2, acc2, half8, lane);

  asm volatile("s_wait_loadcnt 0x0" ::: "memory");  // h gathers must land before residual stores
  int nl = lane & 15, mh = (lane >> 4) * 8;
#pragma unroll
  for (int t = 0; t < 4; ++t) {
    float bv = b2[t * 16 + nl];
#pragma unroll
    for (int r = 0; r < 8; ++r) {
      size_t idx = (size_t)(tile * 16 + mh + r) * HID + t * 16 + nl;
      h[idx] = h[idx] + acc2[t][r] + bv;
    }
  }
}

// decoder: out[N][3] = relu(h@W1+b1)@W2+b2
__global__ void decoder_kernel(const float* __restrict__ h,
                               const _Float16* __restrict__ w1, const float* __restrict__ b1,
                               const _Float16* __restrict__ w2, const float* __restrict__ b2,
                               float* __restrict__ out, int ntiles) {
  __shared__ __align__(32) _Float16 scr[WVB][16 * LSTRIDE];
  int lane = threadIdx.x & 31, wv = threadIdx.x >> 5;
  int tile = blockIdx.x * WVB + wv;
  if (tile >= ntiles) return;
  int m = lane & 15, half8 = (lane >> 4) * 8;
  const float* hp = h + (size_t)(tile * 16 + m) * HID;
  v8f acc[4]; zero4(acc);
  {
    v16h a = a_from_global(hp, half8);
#pragma unroll
    for (int t = 0; t < 4; ++t) acc[t] = wmma16(a, b_from_swz(w1, 0, t, 4, lane), acc[t]);
    v16h b_ = a_from_global(hp + 32, half8);
#pragma unroll
    for (int t = 0; t < 4; ++t) acc[t] = wmma16(b_, b_from_swz(w1, 1, t, 4, lane), acc[t]);
  }
  _Float16* my = scr[wv];
  c_to_lds(my, acc, b1, true, lane);
  const _Float16* myrow = my + (size_t)m * LSTRIDE;
  v8f o = {};
#pragma unroll
  for (int c = 0; c < 2; ++c) {
    v16h a = a_from_lds(myrow, c * 32, half8);
    o = wmma16(a, b_from_swz(w2, c, 0, 1, lane), o);
  }
  int nl = lane & 15, mh = (lane >> 4) * 8;
  if (nl < 3) {
    float bv = b2[nl];
#pragma unroll
    for (int r = 0; r < 8; ++r)
      out[(size_t)(tile * 16 + mh + r) * 3 + nl] = o[r] + bv;
  }
}

// ---------------- host ----------------

static inline size_t align256(size_t x) { return (x + 255) & ~(size_t)255; }

extern "C" void kernel_launch(void* const* d_in, const int* in_sizes, int n_in,
                              void* d_out, int out_size, void* d_ws, size_t ws_size,
                              hipStream_t stream) {
  // --- locate inputs by size signature (robust to pytree flatten order) ---
  int i_x = -1, i_ea = -1, i_u = -1, i_ei = -1, i_b = -1;
  int i_ne = -1, i_ee = -1, i_ge = -1, i_nd = -1;
  int i_le[3], i_n1[3], i_n2[3];
  int ce = 0, c1 = 0, c2 = 0;
  for (int i = 0; i < n_in; ++i) {
    int s = in_sizes[i];
    if (s == NND * 6) i_x = i;
    else if (s == NED * 3) i_ea = i;
    else if (s == NB * 4) i_u = i;
    else if (s == 2 * NED) i_ei = i;
    else if (s == NND) i_b = i;
    else if (s == 6 * HID) i_ne = i;                                   // 384
    else if (s == 3 * HID && i + 1 < n_in && in_sizes[i + 1] == HID) i_ee = i;  // 192,64
    else if (s == 4 * HID) i_ge = i;                                   // 256
    else if (s == HID * HID && i + 3 < n_in && in_sizes[i + 1] == HID &&
             in_sizes[i + 2] == 3 * HID && in_sizes[i + 3] == 3) i_nd = i;      // 4096,64,192,3
    else if (s == 4 * HID * HID) { if (ce < 3) i_le[ce++] = i; }       // 16384
    else if (s == 2 * HID * HID) { if (c1 < 3) i_n1[c1++] = i; }       // 8192
    else if (s == 3 * HID * HID) { if (c2 < 3) i_n2[c2++] = i; }       // 12288
  }
  auto P = [&](int i) { return (const float*)d_in[i]; };
  const float* x = P(i_x);
  const float* ea = P(i_ea);
  const float* u = P(i_u);
  const int* rowI = (const int*)d_in[i_ei];
  const int* colI = rowI + NED;
  const int* batch = (const int*)d_in[i_b];

  // --- workspace carve-up ---
  size_t off = 0;
  auto take = [&](size_t bytes) -> void* {
    void* p = (char*)d_ws + off;
    off = align256(off + bytes);
    return p;
  };
  float* h = (float*)take((size_t)NND * HID * 4);
  float* agg = (float*)take((size_t)NND * HID * 4);
  float* cnt = (float*)take((size_t)NND * 4);
  float* g = (float*)take((size_t)NB * HID * 4);
  float* e = (float*)take((size_t)NED * HID * 4);
  auto takeW = [&](int K, int NT) { return (_Float16*)take((size_t)((K + 31) / 32) * NT * 512 * 2); };
  _Float16* s_ne1 = takeW(6, 4);   _Float16* s_ne2 = takeW(64, 4);
  _Float16* s_ee1 = takeW(3, 4);   _Float16* s_ee2 = takeW(64, 4);
  _Float16* s_nd1 = takeW(64, 4);  _Float16* s_nd2 = takeW(64, 1);
  _Float16 *s_e1[3], *s_e2[3], *s_m1[3], *s_m2[3], *s_x1[3], *s_x2[3];
  for (int l = 0; l < 3; ++l) {
    s_e1[l] = takeW(256, 4); s_e2[l] = takeW(64, 4);
    s_m1[l] = takeW(128, 4); s_m2[l] = takeW(64, 4);
    s_x1[l] = takeW(192, 4); s_x2[l] = takeW(64, 4);
  }
  (void)ws_size; (void)out_size;

  // --- init + prep ---
  zero_kernel<<<4096, BT, 0, stream>>>(agg, (long)NND * HID);
  zero_kernel<<<256, BT, 0, stream>>>(cnt, (long)NND);
  count_kernel<<<(NED + BT - 1) / BT, BT, 0, stream>>>(colI, cnt, NED);

  auto swz = [&](const float* W, _Float16* dst, int K, int Nout, int NT) {
    int total = ((K + 31) / 32) * NT * 512;
    swizzle_w_kernel<<<(total + BT - 1) / BT, BT, 0, stream>>>(W, dst, K, Nout, NT);
  };
  swz(P(i_ne), s_ne1, 6, 64, 4);      swz(P(i_ne + 2), s_ne2, 64, 64, 4);
  swz(P(i_ee), s_ee1, 3, 64, 4);      swz(P(i_ee + 2), s_ee2, 64, 64, 4);
  swz(P(i_nd), s_nd1, 64, 64, 4);     swz(P(i_nd + 2), s_nd2, 64, 3, 1);
  for (int l = 0; l < 3; ++l) {
    swz(P(i_le[l]), s_e1[l], 256, 64, 4); swz(P(i_le[l] + 2), s_e2[l], 64, 64, 4);
    swz(P(i_n1[l]), s_m1[l], 128, 64, 4); swz(P(i_n1[l] + 2), s_m2[l], 64, 64, 4);
    swz(P(i_n2[l]), s_x1[l], 192, 64, 4); swz(P(i_n2[l] + 2), s_x2[l], 64, 64, 4);
  }

  // --- encoders ---
  glob_encoder_kernel<<<1, 64, 0, stream>>>(u, P(i_ge), P(i_ge + 1), P(i_ge + 2), P(i_ge + 3), g);
  int ntilesN = NND / 16, ntilesE = NED / 16;
  encoder_kernel<<<(ntilesN + WVB - 1) / WVB, BT, 0, stream>>>(
      x, ntilesN, 6, s_ne1, P(i_ne + 1), s_ne2, P(i_ne + 3), h);
  encoder_kernel<<<(ntilesE + WVB - 1) / WVB, BT, 0, stream>>>(
      ea, ntilesE, 3, s_ee1, P(i_ee + 1), s_ee2, P(i_ee + 3), e);

  // --- message-passing layers ---
  for (int l = 0; l < 3; ++l) {
    edge_layer_kernel<<<ntilesE / WVB, BT, 0, stream>>>(
        h, e, g, rowI, colI, batch,
        s_e1[l], P(i_le[l] + 1), s_e2[l], P(i_le[l] + 3),
        s_m1[l], P(i_n1[l] + 1), s_m2[l], P(i_n1[l] + 3), agg);
    node_layer_kernel<<<(ntilesN + WVB - 1) / WVB, BT, 0, stream>>>(
        h, agg, cnt, g, batch,
        s_x1[l], P(i_n2[l] + 1), s_x2[l], P(i_n2[l] + 3), ntilesN);
    if (l < 2) zero_kernel<<<4096, BT, 0, stream>>>(agg, (long)NND * HID);
  }

  // --- decoder ---
  decoder_kernel<<<(ntilesN + WVB - 1) / WVB, BT, 0, stream>>>(
      h, s_nd1, P(i_nd + 1), s_nd2, P(i_nd + 3), (float*)d_out, ntilesN);
}
